// ParameterVAEEncoder_31696858645168
// MI455X (gfx1250) — matrix operand order repaired and checked by reading, not compiled, not run on hardware
//
#include <hip/hip_runtime.h>
#include <hip/hip_bf16.h>

// ---------------------------------------------------------------------------
// ParameterVAEEncoder for MI455X (gfx1250, wave32).
// h @ W matmuls use V_WMMA_F32_16X16X32_BF16 with vectorized (b128) fragment
// loads and LDS-staged transposed weights; everything else is bandwidth/
// atomic bound scatter-gather kept in fp32.
// ---------------------------------------------------------------------------

#define HID     64
#define NHEAD   4
#define CH      16
#define NEGSL   0.2f

typedef __attribute__((ext_vector_type(16))) __bf16 v16bf;
typedef __attribute__((ext_vector_type(8)))  __bf16 v8bf;
typedef __attribute__((ext_vector_type(8)))  float  v8f;

static __device__ __forceinline__ unsigned enc_ord(float f) {
  unsigned u = __float_as_uint(f);
  return (u & 0x80000000u) ? ~u : (u | 0x80000000u);
}
static __device__ __forceinline__ float dec_ord(unsigned k) {
  unsigned u = (k & 0x80000000u) ? (k & 0x7FFFFFFFu) : ~k;
  return __uint_as_float(u);
}

// ---- node encoder: h = relu([x, face_emb[ft]] @ node_W + b), stored bf16 ----
__global__ void k_node_encode(const float* __restrict__ x, const int* __restrict__ ft,
                              const float* __restrict__ femb, const float* __restrict__ W,
                              const float* __restrict__ b, __bf16* __restrict__ h, int n) {
  int idx = blockIdx.x * blockDim.x + threadIdx.x;
  if (idx >= n * HID) return;
  int node = idx >> 6, j = idx & 63;
  float acc = b[j];
  const float* xr = x + (long)node * 9;
  #pragma unroll
  for (int f = 0; f < 9; ++f) acc += xr[f] * W[f * HID + j];
  const float* fe = femb + ft[node] * 8;
  #pragma unroll
  for (int f = 0; f < 8; ++f) acc += fe[f] * W[(9 + f) * HID + j];
  acc = acc > 0.0f ? acc : 0.0f;
  h[idx] = (__bf16)acc;
}

// ---- convert gat_lin weights to bf16, TRANSPOSED: Wt[l][col][k] = W[l][k][col]
__global__ void k_convW(const float* __restrict__ W, __bf16* __restrict__ Wt, int L) {
  int idx = blockIdx.x * blockDim.x + threadIdx.x;
  if (idx >= L * HID * HID) return;
  int l = idx / (HID * HID);
  int r = idx - l * HID * HID;
  int k = r >> 6, col = r & 63;
  Wt[l * HID * HID + col * HID + k] = (__bf16)W[idx];
}

// ---- watt[l][k][hh] = sum_c gat_lin_edge[l][k][hh*16+c] * att_e[l][hh][c] ----
__global__ void k_watt(const float* __restrict__ We, const float* __restrict__ ae,
                       float* __restrict__ watt, int L) {
  int idx = blockIdx.x * blockDim.x + threadIdx.x;
  if (idx >= L * HID * NHEAD) return;
  int l = idx / (HID * NHEAD);
  int r = idx - l * HID * NHEAD;
  int k = r >> 2, hh = r & 3;
  float s = 0.0f;
  #pragma unroll
  for (int c = 0; c < CH; ++c)
    s += We[l * HID * HID + k * HID + hh * CH + c] * ae[l * HID + hh * CH + c];
  watt[l * HID * NHEAD + k * NHEAD + hh] = s;
}

// ---- deg + loop_attr accumulation (edge_h recomputed on the fly) ----
__global__ void k_loop_accum(const int* __restrict__ ei, const float* __restrict__ eattr,
                             const float* __restrict__ eW, const float* __restrict__ eb,
                             float* __restrict__ loop_attr, float* __restrict__ deg, int E) {
  int e = blockIdx.x * blockDim.x + threadIdx.x;
  if (e >= E) return;
  int d = ei[E + e];
  float a0 = eattr[e * 2], a1 = eattr[e * 2 + 1];
  atomicAdd(&deg[d], 1.0f);
  float* dst = loop_attr + (long)d * HID;
  #pragma unroll 8
  for (int k = 0; k < HID; ++k) {
    float v = a0 * eW[k] + a1 * eW[HID + k] + eb[k];
    v = v > 0.0f ? v : 0.0f;
    atomicAdd(&dst[k], v);
  }
}

__global__ void k_loop_norm(float* __restrict__ loop_attr, const float* __restrict__ deg, int n) {
  int idx = blockIdx.x * blockDim.x + threadIdx.x;
  if (idx >= n * HID) return;
  float dg = deg[idx >> 6];
  dg = dg < 1.0f ? 1.0f : dg;
  loop_attr[idx] /= dg;
}

// ---- xh = h @ W via WMMA bf16->f32 ----
// One wave per 16-row slab; 8 wmma issues per wave. Weights (transposed,
// Wt[col][k]) staged in LDS once per block. A fragments are two b128 loads,
// B fragments a contiguous 32B LDS read; A and B both use contiguous k = 16*half + s order.
__global__ void k_xh_wmma(const __bf16* __restrict__ h, const __bf16* __restrict__ Wt,
                          float* __restrict__ xh, int n) {
  __shared__ __bf16 sW[HID * HID];
  {
    const unsigned* src = (const unsigned*)Wt;
    unsigned* dst = (unsigned*)sW;
    for (int i = threadIdx.x; i < HID * HID / 2; i += blockDim.x) dst[i] = src[i];
  }
  __syncthreads();

  int lane = threadIdx.x & 31;
  int wave = threadIdx.x >> 5;
  int tile = blockIdx.x * (blockDim.x >> 5) + wave;
  long r0 = (long)tile * 16;
  if (r0 >= n) return;

  int half = lane >> 4;      // K-half selector for A / B
  int m    = lane & 15;
  int coln = m;

  long rr = r0 + m; if (rr >= n) rr = n - 1;   // clamp; stores are guarded
  const __bf16* hrow = h + rr * HID;

  // A tiles (16x32 bf16): lane half h needs K = 16h..16h+15.
  v8bf a0lo = *(const v8bf*)(hrow + 16 * half);
  v8bf a0hi = *(const v8bf*)(hrow + 8 + 16 * half);
  v8bf a1lo = *(const v8bf*)(hrow + 32 + 16 * half);
  v8bf a1hi = *(const v8bf*)(hrow + 40 + 16 * half);
  v16bf a0 = __builtin_shufflevector(a0lo, a0hi, 0,1,2,3,4,5,6,7,8,9,10,11,12,13,14,15);
  v16bf a1 = __builtin_shufflevector(a1lo, a1hi, 0,1,2,3,4,5,6,7,8,9,10,11,12,13,14,15);

  bool full = (r0 + 16 <= (long)n);

  #pragma unroll
  for (int cb = 0; cb < 4; ++cb) {
    // B tiles (32x16 bf16): lane half h needs K = 16h..16h+15 at fixed column.
    const __bf16* wc = sW + (cb * CH + coln) * HID + 16 * half;
    v16bf b0 = *(const v16bf*)(wc);
    v16bf b1 = *(const v16bf*)(wc + 32);

    v8f c = {};
    c = __builtin_amdgcn_wmma_f32_16x16x32_bf16(false, a0, false, b0, (short)0, c, false, false);
    c = __builtin_amdgcn_wmma_f32_16x16x32_bf16(false, a1, false, b1, (short)0, c, false, false);

    // D layout: VGPR r holds M=r (lanes 0-15) / M=r+8 (lanes 16-31).
    float* xcol = xh + (r0 + 8 * half) * HID + cb * CH + coln;
    if (full) {
      #pragma unroll
      for (int r = 0; r < 8; ++r) xcol[(long)r * HID] = c[r];
    } else {
      #pragma unroll
      for (int r = 0; r < 8; ++r) {
        if (r0 + 8 * half + r < n) xcol[(long)r * HID] = c[r];
      }
    }
  }
}

// ---- per-node attention logits from xh ----
__global__ void k_att_sd(const float* __restrict__ xh, const float* __restrict__ as_,
                         const float* __restrict__ ad_, float* __restrict__ al_s,
                         float* __restrict__ al_d, int n) {
  int idx = blockIdx.x * blockDim.x + threadIdx.x;
  if (idx >= n * NHEAD) return;
  int node = idx >> 2, hh = idx & 3;
  const float* xr = xh + (long)node * HID + hh * CH;
  float s = 0.0f, d = 0.0f;
  #pragma unroll
  for (int c = 0; c < CH; ++c) { s += xr[c] * as_[hh * CH + c]; d += xr[c] * ad_[hh * CH + c]; }
  al_s[idx] = s; al_d[idx] = d;
}

// ---- al_e for edges: recompute edge enc, dot with collapsed watt ----
__global__ void k_al_e_edge(const float* __restrict__ eattr, const float* __restrict__ eW,
                            const float* __restrict__ eb, const float* __restrict__ watt,
                            float* __restrict__ al_e, int E) {
  int e = blockIdx.x * blockDim.x + threadIdx.x;
  if (e >= E) return;
  float a0 = eattr[e * 2], a1 = eattr[e * 2 + 1];
  float s0 = 0, s1 = 0, s2 = 0, s3 = 0;
  #pragma unroll 8
  for (int k = 0; k < HID; ++k) {
    float v = a0 * eW[k] + a1 * eW[HID + k] + eb[k];
    v = v > 0.0f ? v : 0.0f;
    const float* w = watt + k * NHEAD;
    s0 += v * w[0]; s1 += v * w[1]; s2 += v * w[2]; s3 += v * w[3];
  }
  al_e[e * 4 + 0] = s0; al_e[e * 4 + 1] = s1; al_e[e * 4 + 2] = s2; al_e[e * 4 + 3] = s3;
}

__global__ void k_al_e_loop(const float* __restrict__ loop_attr, const float* __restrict__ watt,
                            float* __restrict__ al_el, int n) {
  int idx = blockIdx.x * blockDim.x + threadIdx.x;
  if (idx >= n * NHEAD) return;
  int node = idx >> 2, hh = idx & 3;
  const float* lr = loop_attr + (long)node * HID;
  float s = 0.0f;
  #pragma unroll 8
  for (int k = 0; k < HID; ++k) s += lr[k] * watt[k * NHEAD + hh];
  al_el[idx] = s;
}

// ---- softmax pass 1: alpha + segment max (ordered-uint atomicMax) ----
__global__ void k_alpha_edge(const int* __restrict__ ei, const float* __restrict__ al_s,
                             const float* __restrict__ al_d, const float* __restrict__ al_e,
                             float* __restrict__ ebuf, unsigned* __restrict__ amax, int E) {
  int e = blockIdx.x * blockDim.x + threadIdx.x;
  if (e >= E) return;
  int s = ei[e], d = ei[E + e];
  #pragma unroll
  for (int hh = 0; hh < NHEAD; ++hh) {
    float al = al_s[s * 4 + hh] + al_d[d * 4 + hh] + al_e[e * 4 + hh];
    al = al > 0.0f ? al : NEGSL * al;
    ebuf[e * 4 + hh] = al;
    atomicMax(&amax[d * 4 + hh], enc_ord(al));
  }
}

__global__ void k_alpha_loop(const float* __restrict__ al_s, const float* __restrict__ al_d,
                             const float* __restrict__ al_el, float* __restrict__ lbuf,
                             unsigned* __restrict__ amax, int n) {
  int node = blockIdx.x * blockDim.x + threadIdx.x;
  if (node >= n) return;
  #pragma unroll
  for (int hh = 0; hh < NHEAD; ++hh) {
    float al = al_s[node * 4 + hh] + al_d[node * 4 + hh] + al_el[node * 4 + hh];
    al = al > 0.0f ? al : NEGSL * al;
    lbuf[node * 4 + hh] = al;
    atomicMax(&amax[node * 4 + hh], enc_ord(al));
  }
}

// ---- softmax pass 2: exp + denom ----
__global__ void k_exp_edge(const int* __restrict__ ei, float* __restrict__ ebuf,
                           const unsigned* __restrict__ amax, float* __restrict__ denom, int E) {
  int e = blockIdx.x * blockDim.x + threadIdx.x;
  if (e >= E) return;
  int d = ei[E + e];
  #pragma unroll
  for (int hh = 0; hh < NHEAD; ++hh) {
    float ex = __expf(ebuf[e * 4 + hh] - dec_ord(amax[d * 4 + hh]));
    ebuf[e * 4 + hh] = ex;
    atomicAdd(&denom[d * 4 + hh], ex);
  }
}

__global__ void k_exp_loop(float* __restrict__ lbuf, const unsigned* __restrict__ amax,
                           float* __restrict__ denom, int n) {
  int node = blockIdx.x * blockDim.x + threadIdx.x;
  if (node >= n) return;
  #pragma unroll
  for (int hh = 0; hh < NHEAD; ++hh) {
    float ex = __expf(lbuf[node * 4 + hh] - dec_ord(amax[node * 4 + hh]));
    lbuf[node * 4 + hh] = ex;
    atomicAdd(&denom[node * 4 + hh], ex);
  }
}

// ---- pass 3: weighted scatter of xh[src] (numerator; divide at finalize) ----
__global__ void k_agg_edge(const int* __restrict__ ei, const float* __restrict__ ebuf,
                           const float* __restrict__ xh, float* __restrict__ acc, int E) {
  int e = blockIdx.x * blockDim.x + threadIdx.x;
  if (e >= E) return;
  int s = ei[e], d = ei[E + e];
  float w0 = ebuf[e * 4], w1 = ebuf[e * 4 + 1], w2 = ebuf[e * 4 + 2], w3 = ebuf[e * 4 + 3];
  const float* xs = xh + (long)s * HID;
  float* ad = acc + (long)d * HID;
  #pragma unroll
  for (int c = 0; c < CH; ++c) {
    atomicAdd(&ad[c],          xs[c]          * w0);
    atomicAdd(&ad[CH + c],     xs[CH + c]     * w1);
    atomicAdd(&ad[2 * CH + c], xs[2 * CH + c] * w2);
    atomicAdd(&ad[3 * CH + c], xs[3 * CH + c] * w3);
  }
}

__global__ void k_agg_loop(const float* __restrict__ xh, const float* __restrict__ lbuf,
                           float* __restrict__ acc, int n) {
  int idx = blockIdx.x * blockDim.x + threadIdx.x;
  if (idx >= n * HID) return;
  int node = idx >> 6, hh = (idx & 63) >> 4;
  atomicAdd(&acc[idx], xh[idx] * lbuf[node * 4 + hh]);
}

// ---- finalize: divide by denom, add bias, relu, write next-layer bf16 h ----
__global__ void k_final(const float* __restrict__ acc, const float* __restrict__ denom,
                        const float* __restrict__ bias, __bf16* __restrict__ h, int n) {
  int idx = blockIdx.x * blockDim.x + threadIdx.x;
  if (idx >= n * HID) return;
  int node = idx >> 6, j = idx & 63;
  float v = acc[idx] / denom[node * 4 + (j >> 4)] + bias[j];
  v = v > 0.0f ? v : 0.0f;
  h[idx] = (__bf16)v;
}

// ---- masked mean pooling per graph ----
__global__ void k_pool(const __bf16* __restrict__ h, const float* __restrict__ mask,
                       const int* __restrict__ batch, float* __restrict__ psum,
                       float* __restrict__ pcnt, int n) {
  int idx = blockIdx.x * blockDim.x + threadIdx.x;
  if (idx >= n * HID) return;
  int node = idx >> 6, j = idx & 63;
  float mk = mask[node];
  int g = batch[node];
  atomicAdd(&psum[g * HID + j], (float)h[idx] * mk);
  if (j == 0) atomicAdd(&pcnt[g], mk);
}

// ---- heads: mu / logvar ----
__global__ void k_head(const float* __restrict__ psum, const float* __restrict__ pcnt,
                       const float* __restrict__ muW, const float* __restrict__ muB,
                       const float* __restrict__ lvW, const float* __restrict__ lvB,
                       float* __restrict__ out, int G) {
  int idx = blockIdx.x * blockDim.x + threadIdx.x;
  if (idx >= G * 32) return;
  int g = idx >> 5, o = idx & 31;
  float cnt = pcnt[g]; cnt = cnt < 1.0f ? 1.0f : cnt;
  float mu = muB[o], lv = lvB[o];
  #pragma unroll 8
  for (int k = 0; k < HID; ++k) {
    float hp = psum[g * HID + k] / cnt;
    mu += hp * muW[k * 32 + o];
    lv += hp * lvW[k * 32 + o];
  }
  out[g * 32 + o] = mu;
  out[G * 32 + g * 32 + o] = lv;
}

#define CDIV(a, b) (((a) + (b) - 1) / (b))

extern "C" void kernel_launch(void* const* d_in, const int* in_sizes, int n_in,
                              void* d_out, int out_size, void* d_ws, size_t ws_size,
                              hipStream_t stream) {
  const float* x       = (const float*)d_in[0];
  const int*   ft      = (const int*)  d_in[1];
  const int*   ei      = (const int*)  d_in[2];
  const float* eattr   = (const float*)d_in[3];
  const int*   batch   = (const int*)  d_in[4];
  const float* mask    = (const float*)d_in[5];
  const float* femb    = (const float*)d_in[7];
  const float* nodeW   = (const float*)d_in[8];
  const float* nodeB   = (const float*)d_in[9];
  const float* edgeW   = (const float*)d_in[10];
  const float* edgeB   = (const float*)d_in[11];
  const float* gatlin  = (const float*)d_in[12];
  const float* gatline = (const float*)d_in[13];
  const float* attS    = (const float*)d_in[14];
  const float* attD    = (const float*)d_in[15];
  const float* attE    = (const float*)d_in[16];
  const float* gbias   = (const float*)d_in[17];
  const float* muW     = (const float*)d_in[18];
  const float* muB     = (const float*)d_in[19];
  const float* lvW     = (const float*)d_in[20];
  const float* lvB     = (const float*)d_in[21];
  float* out = (float*)d_out;

  const int N = in_sizes[1];
  const int E = in_sizes[3] / 2;
  const int L = in_sizes[17] / HID;
  const int G = out_size / 64;   // 2 * LAT(32)

  // ---- workspace carve-up ----
  char* base = (char*)d_ws;
  size_t off = 0;
  auto carve = [&](size_t bytes) -> char* {
    char* p = base + off;
    off = (off + bytes + 255) & ~(size_t)255;
    return p;
  };
  __bf16*   h_bf   = (__bf16*)  carve((size_t)N * HID * sizeof(__bf16));
  float*    xh     = (float*)   carve((size_t)N * HID * sizeof(float));
  float*    acc    = (float*)   carve((size_t)N * HID * sizeof(float));
  float*    lattr  = (float*)   carve((size_t)N * HID * sizeof(float));
  float*    deg    = (float*)   carve((size_t)N * sizeof(float));
  float*    al_s   = (float*)   carve((size_t)N * NHEAD * sizeof(float));
  float*    al_d   = (float*)   carve((size_t)N * NHEAD * sizeof(float));
  float*    al_el  = (float*)   carve((size_t)N * NHEAD * sizeof(float));
  unsigned* amax   = (unsigned*)carve((size_t)N * NHEAD * sizeof(unsigned));
  float*    denom  = (float*)   carve((size_t)N * NHEAD * sizeof(float));
  float*    lbuf   = (float*)   carve((size_t)N * NHEAD * sizeof(float));
  float*    al_ee  = (float*)   carve((size_t)E * NHEAD * sizeof(float));
  float*    ebuf   = (float*)   carve((size_t)E * NHEAD * sizeof(float));
  __bf16*   Wt     = (__bf16*)  carve((size_t)L * HID * HID * sizeof(__bf16));
  float*    watt   = (float*)   carve((size_t)L * HID * NHEAD * sizeof(float));
  float*    psum   = (float*)   carve((size_t)G * HID * sizeof(float));
  float*    pcnt   = (float*)   carve((size_t)G * sizeof(float));
  (void)ws_size; (void)n_in;

  const int B = 256;

  // ---- layer-invariant precompute ----
  hipMemsetAsync(deg,   0, (size_t)N * sizeof(float), stream);
  hipMemsetAsync(lattr, 0, (size_t)N * HID * sizeof(float), stream);
  k_node_encode<<<CDIV(N * HID, B), B, 0, stream>>>(x, ft, femb, nodeW, nodeB, h_bf, N);
  k_convW<<<CDIV(L * HID * HID, B), B, 0, stream>>>(gatlin, Wt, L);
  k_watt<<<CDIV(L * HID * NHEAD, B), B, 0, stream>>>(gatline, attE, watt, L);
  k_loop_accum<<<CDIV(E, B), B, 0, stream>>>(ei, eattr, edgeW, edgeB, lattr, deg, E);
  k_loop_norm<<<CDIV(N * HID, B), B, 0, stream>>>(lattr, deg, N);

  // ---- GAT layers ----
  for (int l = 0; l < L; ++l) {
    const int tiles = CDIV(N, 16);
    k_xh_wmma<<<CDIV(tiles, 8), B, 0, stream>>>(h_bf, Wt + (size_t)l * HID * HID, xh, N);
    k_att_sd<<<CDIV(N * NHEAD, B), B, 0, stream>>>(xh, attS + l * HID, attD + l * HID, al_s, al_d, N);
    k_al_e_edge<<<CDIV(E, B), B, 0, stream>>>(eattr, edgeW, edgeB, watt + l * HID * NHEAD, al_ee, E);
    k_al_e_loop<<<CDIV(N * NHEAD, B), B, 0, stream>>>(lattr, watt + l * HID * NHEAD, al_el, N);

    hipMemsetAsync(amax,  0, (size_t)N * NHEAD * sizeof(unsigned), stream);
    hipMemsetAsync(denom, 0, (size_t)N * NHEAD * sizeof(float), stream);
    hipMemsetAsync(acc,   0, (size_t)N * HID * sizeof(float), stream);

    k_alpha_edge<<<CDIV(E, B), B, 0, stream>>>(ei, al_s, al_d, al_ee, ebuf, amax, E);
    k_alpha_loop<<<CDIV(N, B), B, 0, stream>>>(al_s, al_d, al_el, lbuf, amax, N);
    k_exp_edge<<<CDIV(E, B), B, 0, stream>>>(ei, ebuf, amax, denom, E);
    k_exp_loop<<<CDIV(N, B), B, 0, stream>>>(lbuf, amax, denom, N);
    k_agg_edge<<<CDIV(E, B), B, 0, stream>>>(ei, ebuf, xh, acc, E);
    k_agg_loop<<<CDIV(N * HID, B), B, 0, stream>>>(xh, lbuf, acc, N);
    k_final<<<CDIV(N * HID, B), B, 0, stream>>>(acc, denom, gbias + l * HID, h_bf, N);
  }

  // ---- pooling + heads ----
  hipMemsetAsync(psum, 0, (size_t)G * HID * sizeof(float), stream);
  hipMemsetAsync(pcnt, 0, (size_t)G * sizeof(float), stream);
  k_pool<<<CDIV(N * HID, B), B, 0, stream>>>(h_bf, mask, batch, psum, pcnt, N);
  k_head<<<CDIV(G * 32, B), B, 0, stream>>>(psum, pcnt, muW, muB, lvW, lvB, out, G);
}
